// Attention_51573967290658
// MI455X (gfx1250) — matrix operand order repaired and checked by reading, not compiled, not run on hardware
//
#include <hip/hip_runtime.h>
#include <hip/hip_bf16.h>

typedef __bf16 bf16;
typedef __attribute__((ext_vector_type(16))) __bf16 bf16x16;
typedef __attribute__((ext_vector_type(8)))  __bf16 bf16x8;
typedef __attribute__((ext_vector_type(8)))  float  f32x8;

#define B_    4
#define N_    2048
#define D_    1024
#define H_    16
#define DH_   64
#define NK_   2049     // 1 null key + 2048
#define KSTEPS_ 65     // 65*32 = 2080 >= 2049; only the last step needs masking

// ---------- helpers ----------
__device__ __forceinline__ bf16x16 ld_frag(const bf16* p0, const bf16* p1) {
  bf16x8 a = *(const bf16x8*)p0;
  bf16x8 b = *(const bf16x8*)p1;
  bf16x16 r;
#pragma unroll
  for (int i = 0; i < 8; ++i) { r[i] = a[i]; r[i + 8] = b[i]; }
  return r;
}

__device__ __forceinline__ f32x8 wmma_bf16(bf16x16 a, bf16x16 b, f32x8 c) {
  // D = A(16x32 bf16) * B(32x16 bf16) + C(16x16 f32)
  return __builtin_amdgcn_wmma_f32_16x16x32_bf16(false, a, false, b, (short)0, c, false, false);
}

__device__ __forceinline__ f32x8 zero8() {
  f32x8 z;
#pragma unroll
  for (int i = 0; i < 8; ++i) z[i] = 0.0f;
  return z;
}

__device__ __forceinline__ bf16x8 zerob8() {
  bf16x8 z;
#pragma unroll
  for (int i = 0; i < 8; ++i) z[i] = (bf16)0.0f;
  return z;
}

// raw v_exp_f32: flushes hugely-negative args to 0, which is exactly softmax semantics
__device__ __forceinline__ float fast_exp2(float x) { return __builtin_amdgcn_exp2f(x); }

// ---------- fp32 -> bf16 cast ----------
__global__ void cast_kernel(const float* __restrict__ s, bf16* __restrict__ d, int n) {
  int i = blockIdx.x * blockDim.x + threadIdx.x;
  int stride = gridDim.x * blockDim.x;
  for (; i < n; i += stride) d[i] = (bf16)s[i];
}

// ---------- LayerNorm (one block per row of 1024) ----------
__global__ void layernorm_kernel(const float* __restrict__ x, const float* __restrict__ gamma,
                                 bf16* __restrict__ xn) {
  __shared__ float red[256];
  const int row = blockIdx.x;
  const float* xr = x + (size_t)row * D_;
  float v[4];
  float s = 0.0f;
#pragma unroll
  for (int i = 0; i < 4; ++i) { v[i] = xr[threadIdx.x + i * 256]; s += v[i]; }
  red[threadIdx.x] = s;
  __syncthreads();
  for (int off = 128; off > 0; off >>= 1) {
    if (threadIdx.x < off) red[threadIdx.x] += red[threadIdx.x + off];
    __syncthreads();
  }
  const float mu = red[0] * (1.0f / D_);
  __syncthreads();
  float s2 = 0.0f;
#pragma unroll
  for (int i = 0; i < 4; ++i) { float d = v[i] - mu; s2 += d * d; }
  red[threadIdx.x] = s2;
  __syncthreads();
  for (int off = 128; off > 0; off >>= 1) {
    if (threadIdx.x < off) red[threadIdx.x] += red[threadIdx.x + off];
    __syncthreads();
  }
  const float rstd = rsqrtf(red[0] * (1.0f / D_) + 1e-5f);
#pragma unroll
  for (int i = 0; i < 4; ++i) {
    int col = threadIdx.x + i * 256;
    xn[(size_t)row * D_ + col] = (bf16)((v[i] - mu) * rstd * gamma[col]);
  }
}

// ---------- WMMA GEMM: C[M,N] = alpha * A[M,K]bf16 @ B[K,N]bf16 ----------
// Block tile 128x128, BK=32, 8 waves in 2x4; double-buffered LDS, 1 barrier/step.
template <bool OUT_BF16>
__global__ void gemm_kernel(const bf16* __restrict__ A, const bf16* __restrict__ Bm,
                            void* __restrict__ Cp, int M, int Nn, int K, float alpha) {
  __shared__ bf16 As[2][128][40];  // [m][k], pitch 40 (pad 8)
  __shared__ bf16 Bs[2][128][40];  // transposed: [n][k]
  const int tid = threadIdx.x;
  const int lane = tid & 31, w = tid >> 5;
  const int r = lane & 15, half = lane >> 4;
  const int h8 = half * 8, kst = half * 16;
  const int wr = w >> 2, wc = w & 3;
  const int m0 = blockIdx.y * 128, n0 = blockIdx.x * 128;
  const int mb = wr * 64, nb = wc * 32;

  f32x8 acc[4][2];
#pragma unroll
  for (int i = 0; i < 4; ++i)
#pragma unroll
    for (int j = 0; j < 2; ++j) acc[i][j] = zero8();

  const int arow = tid >> 1, aseg = (tid & 1) * 16;
  const int bk = tid >> 3, bseg = (tid & 7) * 16;

  // ---- prologue: stage tile 0 into buffer 0 ----
  {
    const bf16* ap = A + (size_t)(m0 + arow) * K + aseg;
    bf16x8 a0 = *(const bf16x8*)ap;
    bf16x8 a1 = *(const bf16x8*)(ap + 8);
    *(bf16x8*)&As[0][arow][aseg] = a0;
    *(bf16x8*)&As[0][arow][aseg + 8] = a1;
    const bf16* bp = Bm + (size_t)bk * Nn + n0 + bseg;
    bf16x8 b0 = *(const bf16x8*)bp;
    bf16x8 b1 = *(const bf16x8*)(bp + 8);
#pragma unroll
    for (int e = 0; e < 8; ++e) {
      Bs[0][bseg + e][bk] = b0[e];
      Bs[0][bseg + 8 + e][bk] = b1[e];
    }
  }
  int cur = 0;

#pragma unroll 1
  for (int kk = 0; kk < K; kk += 32) {
    __syncthreads();
    const bool nxt = (kk + 32) < K;
    bf16x8 na0 = zerob8(), na1 = zerob8(), nb0 = zerob8(), nb1 = zerob8();
    if (nxt) {
      const bf16* ap = A + (size_t)(m0 + arow) * K + (kk + 32) + aseg;
      na0 = *(const bf16x8*)ap;
      na1 = *(const bf16x8*)(ap + 8);
      const bf16* bp = Bm + (size_t)(kk + 32 + bk) * Nn + n0 + bseg;
      nb0 = *(const bf16x8*)bp;
      nb1 = *(const bf16x8*)(bp + 8);
      if (kk + 64 < K) {
        __builtin_prefetch(A + (size_t)(m0 + arow) * K + (kk + 64) + aseg, 0, 3);
        __builtin_prefetch(Bm + (size_t)(kk + 64 + bk) * Nn + n0 + bseg, 0, 3);
      }
    }

    // compute from current buffer
    bf16x16 af[4], bfv[2];
#pragma unroll
    for (int mi = 0; mi < 4; ++mi) {
      const bf16* p = &As[cur][mb + mi * 16 + r][h8];  // A frag: K {h8..h8+7, 16+h8..}
      af[mi] = ld_frag(p, p + 16);
    }
#pragma unroll
    for (int ni = 0; ni < 2; ++ni) {
      const bf16* p = &Bs[cur][nb + ni * 16 + r][h8]; // B frag: K {h8..h8+7, 16+h8..}
      bfv[ni] = ld_frag(p, p + 16);
    }
#pragma unroll
    for (int mi = 0; mi < 4; ++mi)
#pragma unroll
      for (int ni = 0; ni < 2; ++ni)
        acc[mi][ni] = wmma_bf16(af[mi], bfv[ni], acc[mi][ni]);

    // stage next tile into other buffer
    if (nxt) {
      *(bf16x8*)&As[cur ^ 1][arow][aseg] = na0;
      *(bf16x8*)&As[cur ^ 1][arow][aseg + 8] = na1;
#pragma unroll
      for (int e = 0; e < 8; ++e) {
        Bs[cur ^ 1][bseg + e][bk] = nb0[e];
        Bs[cur ^ 1][bseg + 8 + e][bk] = nb1[e];
      }
    }
    cur ^= 1;
  }

  // epilogue: C/D layout -> M = i + 8*half, N = r
#pragma unroll
  for (int mi = 0; mi < 4; ++mi)
#pragma unroll
    for (int ni = 0; ni < 2; ++ni)
#pragma unroll
      for (int i = 0; i < 8; ++i) {
        const int row = m0 + mb + mi * 16 + h8 + i;
        const int col = n0 + nb + ni * 16 + r;
        const float vv = acc[mi][ni][i] * alpha;
        if (OUT_BF16)
          ((bf16*)Cp)[(size_t)row * Nn + col] = (bf16)vv;
        else
          ((float*)Cp)[(size_t)row * Nn + col] = vv;
      }
}

// ---------- Flash attention step (S^T formulation) ----------
// Scores arrive pre-scaled by log2(e) (folded into q), so exponentials are raw v_exp_f32.
// Only the final tile needs key masking -> MASK template, mask-free steady state.
template <bool MASK>
__device__ __forceinline__ void attn_step(const bf16 (*Ksb)[72], const bf16 (*Vsb)[40],
                                          const bf16x16* qb, f32x8* of,
                                          float& mrow, float& lrow,
                                          int keybase, int r, int h8, int kst) {
  // S^T = K_tile @ Q^T : 2 key sub-tiles (A frags) x 2 dim chunks
  f32x8 s0 = zero8(), s1 = zero8();
#pragma unroll
  for (int c = 0; c < 2; ++c) {
    const bf16* ka = &Ksb[r][c * 32 + h8];        // keys 0..15 of tile
    const bf16* kb2 = &Ksb[16 + r][c * 32 + h8];  // keys 16..31 of tile
    s0 = wmma_bf16(ld_frag(ka, ka + 16), qb[c], s0);
    s1 = wmma_bf16(ld_frag(kb2, kb2 + 16), qb[c], s1);
  }

  // online softmax: per-lane (query r); keys live in VGPR elements
  // s0 vgpr i -> key keybase + 8*half + i ; s1 vgpr i -> key keybase + 16 + 8*half + i
  float a0v[8], a1v[8];
  float mx = -1e30f;
#pragma unroll
  for (int i = 0; i < 8; ++i) {
    float a0 = s0[i], a1 = s1[i];
    if (MASK) {
      a0 = (keybase + h8 + i < NK_) ? a0 : -1e30f;
      a1 = (keybase + 16 + h8 + i < NK_) ? a1 : -1e30f;
    }
    a0v[i] = a0; a1v[i] = a1;
    mx = fmaxf(mx, fmaxf(a0, a1));
  }
  mx = fmaxf(mx, __shfl_xor(mx, 16, 32));  // combine the two half-wave key sets
  const float mn = fmaxf(mrow, mx);
  const float corr = fast_exp2(mrow - mn); // scores are in log2 domain
  mrow = mn;

  float ps = 0.0f;
  bf16x16 pf;  // exp2(P) packs directly into the A-fragment layout for P@V
#pragma unroll
  for (int i = 0; i < 8; ++i) {
    const float e0 = fast_exp2(a0v[i] - mn);
    const float e1 = fast_exp2(a1v[i] - mn);
    ps += e0 + e1;
    pf[i] = (bf16)e0;
    pf[i + 8] = (bf16)e1;
  }
  ps += __shfl_xor(ps, 16, 32);
  lrow = lrow * corr + ps;

  // rescale O: of vgpr i holds row (i + 8*half); fetch that row's corr from its lane
#pragma unroll
  for (int i = 0; i < 8; ++i) {
    const float cb = __shfl(corr, h8 + i, 16);
#pragma unroll
    for (int cc = 0; cc < 4; ++cc) of[cc][i] *= cb;
  }

  // O += P @ V
#pragma unroll
  for (int cc = 0; cc < 4; ++cc) {
    const bf16* vp = &Vsb[cc * 16 + r][h8];
    of[cc] = wmma_bf16(pf, ld_frag(vp, vp + 16), of[cc]);
  }
}

// grid: (16 query-tiles, 64 b*h); block 256 = 8 waves, each wave owns 16 query rows.
__global__ void attn_kernel(const bf16* __restrict__ q, const bf16* __restrict__ kv,
                            const float* __restrict__ nullkv, bf16* __restrict__ attn_out) {
  __shared__ bf16 Ks[2][32][72];  // [key][dim], pad 8
  __shared__ bf16 Vs[2][64][40];  // transposed [dim][key], pad 8
  const int tid = threadIdx.x, lane = tid & 31, w = tid >> 5;
  const int r = lane & 15, half = lane >> 4;
  const int h8 = half * 8, kst = half * 16;
  const int mt = blockIdx.x, bh = blockIdx.y;
  const int b = bh >> 4, h = bh & 15;
  const int m0 = mt * 128 + w * 16;
  const int skey = tid >> 3, sseg = (tid & 7) * 8;  // staging: 8 threads per key

  // Q fragments in B-layout (columns of Q^T): lane = query r, dims {h8..h8+7, 16+h8..} per half
  const bf16* qrow = q + ((size_t)(b * N_ + m0 + r)) * D_ + h * DH_;
  bf16x16 qb[2];
#pragma unroll
  for (int c = 0; c < 2; ++c)
    qb[c] = ld_frag(qrow + c * 32 + h8, qrow + c * 32 + h8 + 16);

  f32x8 of[4];
#pragma unroll
  for (int cc = 0; cc < 4; ++cc) of[cc] = zero8();
  float mrow = -1e30f, lrow = 0.0f;  // per-lane stats for query r (log2 domain)

  // ---- prologue: stage key tile 0 (contains the null key at j==0) ----
  {
    bf16x8 kvec, vvec;
    if (skey == 0) {
#pragma unroll
      for (int e = 0; e < 8; ++e) {
        kvec[e] = (bf16)nullkv[(0 * H_ + h) * DH_ + sseg + e];
        vvec[e] = (bf16)nullkv[(1 * H_ + h) * DH_ + sseg + e];
      }
    } else {
      const bf16* p = kv + ((size_t)(b * N_ + skey - 1)) * (2 * D_) + h * DH_ + sseg;
      kvec = *(const bf16x8*)p;
      vvec = *(const bf16x8*)(p + D_);
    }
    *(bf16x8*)&Ks[0][skey][sseg] = kvec;
#pragma unroll
    for (int e = 0; e < 8; ++e) Vs[0][sseg + e][skey] = vvec[e];
  }
  int cur = 0;

  // ---- steady state: 64 mask-free steps, double-buffered ----
#pragma unroll 1
  for (int jt = 0; jt < KSTEPS_ - 1; ++jt) {
    const int keybase = jt * 32;
    __syncthreads();

    // issue next tile's global loads early (never the null key)
    bf16x8 nk = zerob8(), nv = zerob8();
    {
      const int j = keybase + 32 + skey;
      if (j <= N_) {
        const bf16* p = kv + ((size_t)(b * N_ + j - 1)) * (2 * D_) + h * DH_ + sseg;
        nk = *(const bf16x8*)p;
        nv = *(const bf16x8*)(p + D_);
      }
      const int j2 = keybase + 64 + skey;
      if (j2 <= N_)
        __builtin_prefetch(kv + ((size_t)(b * N_ + j2 - 1)) * (2 * D_) + h * DH_ + sseg, 0, 3);
    }

    attn_step<false>(Ks[cur], Vs[cur], qb, of, mrow, lrow, keybase, r, h8, kst);

    // stage next tile into the other buffer
    *(bf16x8*)&Ks[cur ^ 1][skey][sseg] = nk;
#pragma unroll
    for (int e = 0; e < 8; ++e) Vs[cur ^ 1][sseg + e][skey] = nv[e];
    cur ^= 1;
  }

  // ---- masked tail step (keys 2048..2079; only 2048 valid) ----
  __syncthreads();
  attn_step<true>(Ks[cur], Vs[cur], qb, of, mrow, lrow, (KSTEPS_ - 1) * 32, r, h8, kst);

  // ---- normalize + write out in [b, n, h*64] bf16 for the output GEMM ----
  // lrow >= 1 (max element contributes 2^0), so raw v_rcp_f32 is safe here.
  const float inv = __builtin_amdgcn_rcpf(lrow);
#pragma unroll
  for (int i = 0; i < 8; ++i) {
    const float ib = __shfl(inv, h8 + i, 16);
    const int row = m0 + h8 + i;
#pragma unroll
    for (int cc = 0; cc < 4; ++cc)
      attn_out[((size_t)(b * N_ + row)) * D_ + h * DH_ + cc * 16 + r] =
          (bf16)(of[cc][i] * ib);
  }
}

// ---------- host ----------
extern "C" void kernel_launch(void* const* d_in, const int* in_sizes, int n_in,
                              void* d_out, int out_size, void* d_ws, size_t ws_size,
                              hipStream_t stream) {
  (void)in_sizes; (void)n_in; (void)out_size; (void)ws_size;
  const float* x       = (const float*)d_in[0];
  const float* gamma   = (const float*)d_in[1];
  const float* w_q     = (const float*)d_in[2];
  const float* w_kv    = (const float*)d_in[3];
  const float* w_out   = (const float*)d_in[4];
  const float* null_kv = (const float*)d_in[5];
  float* out = (float*)d_out;

  char* ws = (char*)d_ws;
  const size_t MB = 1024 * 1024;
  bf16* wq_b  = (bf16*)(ws + 0 * MB);   //  2 MB
  bf16* wkv_b = (bf16*)(ws + 2 * MB);   //  4 MB
  bf16* wo_b  = (bf16*)(ws + 6 * MB);   //  2 MB
  bf16* xn    = (bf16*)(ws + 8 * MB);   // 16 MB
  bf16* qb    = (bf16*)(ws + 24 * MB);  // 16 MB (scaled by log2(e)/sqrt(64))
  bf16* kvb   = (bf16*)(ws + 40 * MB);  // 32 MB ([row][k|v]: cols 0..1023 k, 1024..2047 v)
  bf16* attnb = (bf16*)(ws + 72 * MB);  // 16 MB

  cast_kernel<<<2048, 256, 0, stream>>>(w_q, wq_b, D_ * D_);
  cast_kernel<<<2048, 256, 0, stream>>>(w_kv, wkv_b, D_ * 2 * D_);
  cast_kernel<<<2048, 256, 0, stream>>>(w_out, wo_b, D_ * D_);

  layernorm_kernel<<<B_ * N_, 256, 0, stream>>>(x, gamma, xn);

  // q = xn @ w_q * (scale * log2e)  -> scores land in the log2 domain for exp2
  const float QALPHA = 0.125f * 1.44269504088896340736f;
  gemm_kernel<true><<<dim3(D_ / 128, (B_ * N_) / 128), 256, 0, stream>>>(
      xn, wq_b, qb, B_ * N_, D_, D_, QALPHA);
  gemm_kernel<true><<<dim3((2 * D_) / 128, (B_ * N_) / 128), 256, 0, stream>>>(
      xn, wkv_b, kvb, B_ * N_, 2 * D_, D_, 1.0f);

  attn_kernel<<<dim3(N_ / 128, B_ * H_), 256, 0, stream>>>(qb, kvb, null_kv, attnb);

  // out = attn @ w_out (fp32 epilogue straight to d_out)
  gemm_kernel<false><<<dim3(D_ / 128, (B_ * N_) / 128), 256, 0, stream>>>(
      attnb, wo_b, out, B_ * N_, D_, D_, 1.0f);
}
